// FastSpeech_20718922236416
// MI455X (gfx1250) — matrix operand order repaired and checked by reading, not compiled, not run on hardware
//
#include <hip/hip_runtime.h>
#include <hip/hip_bf16.h>

typedef __attribute__((ext_vector_type(16))) _Float16 v16h;
typedef __attribute__((ext_vector_type(8)))  _Float16 v8h;
typedef __attribute__((ext_vector_type(2)))  _Float16 v2h;
typedef __attribute__((ext_vector_type(8)))  float    v8f;

#define D_MODEL 384
#define NHEAD   4
#define DKH     96
#define HDIM    384   // NHEAD*DKH
#define CHID    1536
#define BATCH   16
#define LPTOK   256
#define TMELS   1024
#define NMELS   80

// ---------------------------------------------------------------------------
// Generic WMMA GEMM: C[z] = act( A[z] * B[z] + bias + residual ), fp32 in/out,
// f16 multiply / f32 accumulate via v_wmma_f32_16x16x32_f16.
// A element (m,k)   -> A[z*a_bs2 + sr*a_rs + k]  (sr = m + a_shift, masked to
//                      the [0,seq_len) window of m's sequence for SAME conv)
// B element (k,n)   -> B[z*b_bs2 + k*b_sk + n*b_sn]
// C element (m,n)   -> C[z*c_bs2 + m*c_rs + n*c_cs]
// Block = 256 thr = 8 waves (2 M x 4 N), block tile 64(M) x 128(N).
// Each wave owns a 32x32 output tile: 2 A frags x 2 B frags -> 4 WMMAs/k-step.
// B tile stored TRANSPOSED in LDS so B fragments are contiguous ds_load_b128.
// Global->reg staging is software-pipelined one k-tile ahead; each thread owns
// a pair of adjacent k columns so A stages as b64 loads and all LDS commits
// are packed 2xf16 b32 stores. All K/strides here are even (required).
// ---------------------------------------------------------------------------
__global__ __launch_bounds__(256) void gemm_wmma_kernel(
    const float* __restrict__ A, const float* __restrict__ Bm,
    const float* __restrict__ bias, const float* __restrict__ resid,
    float* __restrict__ C,
    int M, int N, int K,
    long long a_rs, long long a_bs2,
    long long b_sk, long long b_sn, long long b_bs2,
    long long c_rs, long long c_cs, long long c_bs2,
    long long bias_bs2,
    int a_shift, int seq_len,
    int accumulate, int act)
{
  __shared__ __align__(16) _Float16 As [64 * 40];   // [m][k], stride 40
  __shared__ __align__(16) _Float16 BsT[128 * 40];  // [n][k], stride 40

  const int tid  = threadIdx.x;
  const int lane = tid & 31;
  const int wv   = tid >> 5;   // 0..7
  const int wm   = wv >> 2;    // 0..1  -> 32-row slab
  const int wn   = wv & 3;     // 0..3  -> 32-col slab

  const long long z = blockIdx.z;
  const float* Ab = A  + z * a_bs2;
  const float* Bb = Bm + z * b_bs2;
  const float* biasb = bias ? bias + z * bias_bs2 : nullptr;
  const float* Rb = resid ? resid + z * c_bs2 : nullptr;
  float* Cb = C + z * c_bs2;

  const int m0 = blockIdx.y * 64;
  const int n0 = blockIdx.x * 128;
  const int wm0 = m0 + wm * 32;
  const int wn0 = n0 + wn * 32;

  // ---- per-thread loop-invariant load descriptors ----
  // Each thread owns k-column pair lcol2 (fixed) and rows lrow2 + 16i.
  const int lrow2 = tid >> 4;        // 0..15
  const int lcol2 = (tid & 15) * 2;  // even k column
  const int iars = (int)a_rs;        // all strides fit in 32 bits here
  const int ibsk = (int)b_sk;
  const int ibsn = (int)b_sn;

  int  a_off[4];  bool a_okm[4];
  for (int i = 0; i < 4; ++i) {
    int gm = m0 + lrow2 + i * 16;
    bool ok = gm < M;
    int sr = gm;
    if (seq_len) {
      int t = gm % seq_len;
      int ts = t + a_shift;
      ok = ok && (ts >= 0 && ts < seq_len);
      sr = gm + a_shift;
    }
    a_okm[i] = ok;
    a_off[i] = ok ? sr * iars : 0;
  }
  int  b_off[8]; bool b_okn[8];
  for (int i = 0; i < 8; ++i) {
    int gn = n0 + lrow2 + i * 16;
    bool ok = gn < N;
    b_okn[i] = ok;
    b_off[i] = ok ? gn * ibsn : 0;
  }

  // C fragment layout: VGPR r holds (M = r + 8*(lane>=16), N = lane&15)
  const int frow = (lane >> 4) * 8;
  const int fcol = lane & 15;

  v8f acc[4];  // [mi*2+ni]
  for (int t4 = 0; t4 < 4; ++t4) {
    int mi = t4 >> 1, ni = t4 & 1;
    int gmb = wm0 + mi * 16 + frow;
    int gn  = wn0 + ni * 16 + fcol;
    for (int r = 0; r < 8; ++r) {
      float v = 0.f;
      if (accumulate && (gmb + r) < M && gn < N)
        v = Cb[(long long)(gmb + r) * c_rs + (long long)gn * c_cs];
      acc[t4][r] = v;
    }
  }

  float rAx[4], rAy[4], rBx[8], rBy[8];
  auto fetch = [&](int k0) {
    int gk  = k0 + lcol2;
    bool okk = gk < K;                 // K even -> gk+1 also valid
    int gkc = okk ? gk : 0;
    for (int i = 0; i < 4; ++i) {
      float2 v = *(const float2*)&Ab[a_off[i] + gkc];   // 8B aligned
      bool ok = okk && a_okm[i];
      rAx[i] = ok ? v.x : 0.f;
      rAy[i] = ok ? v.y : 0.f;
    }
    int r0 = gkc * ibsk;
    int r1 = okk ? (gk + 1) * ibsk : 0;
    for (int i = 0; i < 8; ++i) {
      float vx = Bb[r0 + b_off[i]];
      float vy = Bb[r1 + b_off[i]];
      bool ok = okk && b_okn[i];
      rBx[i] = ok ? vx : 0.f;
      rBy[i] = ok ? vy : 0.f;
    }
  };

  const int ktiles = (K + 31) >> 5;
  fetch(0);                                      // prologue
  for (int kb = 0; kb < ktiles; ++kb) {
    // ---- commit staged regs to LDS (f32 -> packed f16 pairs, b32 stores) ----
    for (int i = 0; i < 4; ++i) {
      v2h p; p[0] = (_Float16)rAx[i]; p[1] = (_Float16)rAy[i];
      *(v2h*)&As[(lrow2 + i * 16) * 40 + lcol2] = p;
    }
    for (int i = 0; i < 8; ++i) {
      v2h p; p[0] = (_Float16)rBx[i]; p[1] = (_Float16)rBy[i];
      *(v2h*)&BsT[(lrow2 + i * 16) * 40 + lcol2] = p;
    }
    __syncthreads();

    // ---- prefetch next tile while WMMAs run ----
    if (kb + 1 < ktiles) fetch((kb + 1) << 5);

    // ---- build fragments per ISA 7.12.2 ----
    // A 16x32: lanes 0-15 (kh=0): K 0-7 then 16-23; lanes 16-31 (kh=8): K 8-15 / 24-31
    // B 32x16: lane holds column n; same K order as A: K kbse..kbse+7 then 16+kbse..16+kbse+7
    v16h af[2], bf[2];
    const int kh   = (lane >> 4) * 8;
    const int kbse = (lane >> 4) * 8;
    for (int mi = 0; mi < 2; ++mi) {
      int ml = wm * 32 + mi * 16 + (lane & 15);
      v8h a0 = *(const v8h*)&As[ml * 40 + kh];
      v8h a1 = *(const v8h*)&As[ml * 40 + 16 + kh];
      af[mi] = __builtin_shufflevector(a0, a1, 0,1,2,3,4,5,6,7,8,9,10,11,12,13,14,15);
    }
    for (int ni = 0; ni < 2; ++ni) {
      int nl = wn * 32 + ni * 16 + (lane & 15);
      v8h b0 = *(const v8h*)&BsT[nl * 40 + kbse];
      v8h b1 = *(const v8h*)&BsT[nl * 40 + kbse + 16];
      bf[ni] = __builtin_shufflevector(b0, b1, 0,1,2,3,4,5,6,7,8,9,10,11,12,13,14,15);
    }
    acc[0] = __builtin_amdgcn_wmma_f32_16x16x32_f16(false, af[0], false, bf[0], (short)0, acc[0], false, false);
    acc[1] = __builtin_amdgcn_wmma_f32_16x16x32_f16(false, af[0], false, bf[1], (short)0, acc[1], false, false);
    acc[2] = __builtin_amdgcn_wmma_f32_16x16x32_f16(false, af[1], false, bf[0], (short)0, acc[2], false, false);
    acc[3] = __builtin_amdgcn_wmma_f32_16x16x32_f16(false, af[1], false, bf[1], (short)0, acc[3], false, false);
    __syncthreads();
  }

  // ---- epilogue: bias + residual + activation, strided store ----
  for (int t4 = 0; t4 < 4; ++t4) {
    int mi = t4 >> 1, ni = t4 & 1;
    int gmb = wm0 + mi * 16 + frow;
    int gn  = wn0 + ni * 16 + fcol;
    if (gn >= N) continue;
    float bv = biasb ? biasb[gn] : 0.f;
    for (int r = 0; r < 8; ++r) {
      int gm = gmb + r;
      if (gm < M) {
        float v = acc[t4][r] + bv;
        if (Rb) v += Rb[(long long)gm * c_rs + (long long)gn * c_cs];
        if (act == 1) v = fmaxf(v, 0.f);
        Cb[(long long)gm * c_rs + (long long)gn * c_cs] = v;
      }
    }
  }
}

// ---------------------------------------------------------------------------
// LayerNorm over last dim (D<=384, multiple of 32). One wave per row.
// Optional residual input: y = LN(x + r).
// ---------------------------------------------------------------------------
__global__ __launch_bounds__(256) void ln_kernel(
    const float* __restrict__ x, const float* __restrict__ resid,
    const float* __restrict__ g, const float* __restrict__ beta,
    float* __restrict__ y, int rows, int D)
{
  int lane = threadIdx.x & 31;
  int row = blockIdx.x * 8 + (threadIdx.x >> 5);
  if (row >= rows) return;
  const float* xr = x + (long long)row * D;
  const float* rr = resid ? resid + (long long)row * D : nullptr;
  float vals[12];
  int n = D / 32;
  float s = 0.f;
  for (int i = 0; i < n; ++i) {
    float v = xr[lane + i * 32];
    if (rr) v += rr[lane + i * 32];
    vals[i] = v; s += v;
  }
  for (int o = 16; o; o >>= 1) s += __shfl_xor(s, o, 32);
  float mean = s / (float)D;
  float vs = 0.f;
  for (int i = 0; i < n; ++i) { float d = vals[i] - mean; vs += d * d; }
  for (int o = 16; o; o >>= 1) vs += __shfl_xor(vs, o, 32);
  float inv = rsqrtf(vs / (float)D + 1e-5f);
  float* yr = y + (long long)row * D;
  for (int i = 0; i < n; ++i) {
    int c = lane + i * 32;
    yr[c] = (vals[i] - mean) * inv * g[c] + beta[c];
  }
}

// ---------------------------------------------------------------------------
// Row softmax with scale folded in: softmax(scale * x). One block per row.
// ---------------------------------------------------------------------------
__global__ __launch_bounds__(256) void softmax_kernel(
    float* __restrict__ S, int T, float scale)
{
  long long row = blockIdx.x;
  float* r = S + row * (long long)T;
  __shared__ float red[256];
  int tid = threadIdx.x;
  float m = -3.4e38f;
  for (int i = tid; i < T; i += 256) m = fmaxf(m, r[i]);
  red[tid] = m; __syncthreads();
  for (int o = 128; o; o >>= 1) { if (tid < o) red[tid] = fmaxf(red[tid], red[tid + o]); __syncthreads(); }
  m = red[0]; __syncthreads();
  float s = 0.f;
  for (int i = tid; i < T; i += 256) { float e = __expf((r[i] - m) * scale); r[i] = e; s += e; }
  red[tid] = s; __syncthreads();
  for (int o = 128; o; o >>= 1) { if (tid < o) red[tid] += red[tid + o]; __syncthreads(); }
  float invs = 1.f / red[0];
  for (int i = tid; i < T; i += 256) r[i] *= invs;
}

// ---------------------------------------------------------------------------
// x[b,t,:] = emb[tokens[b,t],:] * sqrt(D) + PE(t,:)
// ---------------------------------------------------------------------------
__global__ void embed_pe_kernel(const int* __restrict__ tokens,
                                const float* __restrict__ emb,
                                float* __restrict__ x, int L, int D)
{
  int bt = blockIdx.x;
  int t = bt % L;
  int tok = tokens[bt];
  for (int d = threadIdx.x; d < D; d += blockDim.x) {
    int i = d >> 1;
    float div = __expf((float)(2 * i) / (float)D * -9.210340371976184f); // -ln 1e4
    float ang = (float)t * div;
    float pe = (d & 1) ? __cosf(ang) : __sinf(ang);
    x[(long long)bt * D + d] = emb[(long long)tok * D + d] * 19.595917942265423f + pe;
  }
}

// ---------------------------------------------------------------------------
// Length regulation: LDS scan of masked durations, searchsorted-right, gather
// rows, mask past total, add PE(TMELS). One block (256 thr) per batch.
// ---------------------------------------------------------------------------
__global__ __launch_bounds__(256) void length_reg_kernel(
    const float* __restrict__ x, const int* __restrict__ durations,
    const int* __restrict__ tlens, float* __restrict__ mel,
    int LPc, int TM, int D)
{
  int b = blockIdx.x;
  __shared__ int ends[256];
  __shared__ int idx[1024];
  int tid = threadIdx.x;
  int tl = tlens[b];
  ends[tid] = (tid < tl) ? durations[b * LPc + tid] : 0;
  __syncthreads();
  for (int off = 1; off < 256; off <<= 1) {
    int v = (tid >= off) ? ends[tid - off] : 0;
    __syncthreads();
    ends[tid] += v;
    __syncthreads();
  }
  int total = ends[255];
  for (int t = tid; t < TM; t += 256) {
    int lo = 0, hi = LPc;
    while (lo < hi) { int mid = (lo + hi) >> 1; if (ends[mid] <= t) lo = mid + 1; else hi = mid; }
    idx[t] = (lo < LPc - 1) ? lo : (LPc - 1);
  }
  __syncthreads();
  long long nel = (long long)TM * D;
  for (long long e = tid; e < nel; e += 256) {
    int t  = (int)(e / D);
    int dd = (int)(e % D);
    float v = 0.f;
    if (t < total) v = x[((long long)b * LPc + idx[t]) * D + dd];
    int i = dd >> 1;
    float div = __expf((float)(2 * i) / (float)D * -9.210340371976184f);
    float ang = (float)t * div;
    float pe = (dd & 1) ? __cosf(ang) : __sinf(ang);
    mel[((long long)b * TM + t) * D + dd] = v + pe;
  }
}

// ---------------------------------------------------------------------------
// Host orchestration
// ---------------------------------------------------------------------------
extern "C" void kernel_launch(void* const* d_in, const int* in_sizes, int n_in,
                              void* d_out, int out_size, void* d_ws, size_t ws_size,
                              hipStream_t stream) {
  (void)in_sizes; (void)n_in; (void)out_size; (void)ws_size;

  const int*   tokens = (const int*)d_in[0];
  const int*   tlens  = (const int*)d_in[1];
  const int*   durs   = (const int*)d_in[2];
  const float* emb    = (const float*)d_in[3];
  auto P = [&](int i) { return (const float*)d_in[i]; };
  // input index map: 4 + 6*18 enc, 6*18 dec, 10 dp (unused), out_w, out_b
  const int ENC0 = 4, DEC0 = 4 + 6 * 18, OWI = 4 + 12 * 18 + 10, OBI = OWI + 1;

  // workspace layout (floats)
  float* ws = (float*)d_ws;
  const long long SZ_BT   = (long long)BATCH * TMELS * D_MODEL;  // 6.29M
  float* x1   = ws;                 // post-LN scratch
  float* qb   = x1  + SZ_BT;
  float* kb   = qb  + SZ_BT;
  float* vb   = kb  + SZ_BT;
  float* zb   = vb  + SZ_BT;
  float* x3   = zb  + SZ_BT;
  float* h2   = x3  + SZ_BT;
  float* mel  = h2  + SZ_BT;
  float* hbuf = mel + SZ_BT;                                   // [BT, CHID]
  float* Sbuf = hbuf + (long long)BATCH * TMELS * CHID;        // [H, T, T]
  float* xe   = Sbuf + (long long)NHEAD * TMELS * TMELS;       // [B, LP, D]

  auto gemm = [&](const float* A, const float* B, const float* bias,
                  const float* resid, float* C,
                  int M, int N, int K,
                  long long a_rs, long long a_bs2,
                  long long b_sk, long long b_sn, long long b_bs2,
                  long long c_rs, long long c_cs, long long c_bs2,
                  long long bias_bs2, int a_shift, int seqlen,
                  int accum, int act, int Z) {
    dim3 g((N + 127) / 128, (M + 63) / 64, Z);
    gemm_wmma_kernel<<<g, 256, 0, stream>>>(A, B, bias, resid, C, M, N, K,
        a_rs, a_bs2, b_sk, b_sn, b_bs2, c_rs, c_cs, c_bs2, bias_bs2,
        a_shift, seqlen, accum, act);
  };

  auto fft_block = [&](float* x, int Bn, int T, int pib) {
    const int BT = Bn * T;
    const float *ln1g=P(pib+0), *ln1b=P(pib+1), *Wq=P(pib+2), *bq=P(pib+3),
      *Wk=P(pib+4), *bk=P(pib+5), *Wv=P(pib+6), *bv=P(pib+7), *Wo=P(pib+8),
      *bo=P(pib+9), *ln2g=P(pib+10), *ln2b=P(pib+11), *c1w=P(pib+12),
      *c1b=P(pib+13), *c2w=P(pib+14), *c2b=P(pib+15), *ln3g=P(pib+16),
      *ln3b=P(pib+17);

    // x1 = LN(x)
    ln_kernel<<<(BT + 7) / 8, 256, 0, stream>>>(x, nullptr, ln1g, ln1b, x1, BT, D_MODEL);
    // Q/K/V: per-head batched (z = h). Wq[h] is [D, DK] row-major.
    gemm(x1, Wq, bq, nullptr, qb, BT, DKH, D_MODEL, D_MODEL, 0,
         DKH, 1, (long long)D_MODEL * DKH, HDIM, 1, DKH, DKH, 0, 0, 0, 0, NHEAD);
    gemm(x1, Wk, bk, nullptr, kb, BT, DKH, D_MODEL, D_MODEL, 0,
         DKH, 1, (long long)D_MODEL * DKH, HDIM, 1, DKH, DKH, 0, 0, 0, 0, NHEAD);
    gemm(x1, Wv, bv, nullptr, vb, BT, DKH, D_MODEL, D_MODEL, 0,
         DKH, 1, (long long)D_MODEL * DKH, HDIM, 1, DKH, DKH, 0, 0, 0, 0, NHEAD);
    // attention, per batch (Sbuf holds one batch's H*T*T scores)
    for (int b = 0; b < Bn; ++b) {
      const float* qp = qb + (long long)b * T * HDIM;
      const float* kp = kb + (long long)b * T * HDIM;
      const float* vp = vb + (long long)b * T * HDIM;
      float* zp = zb + (long long)b * T * HDIM;
      // S[h] = Q[h] * K[h]^T   (B addressed as (k,n) -> kp[n*HDIM + k])
      gemm(qp, kp, nullptr, nullptr, Sbuf, T, T, DKH,
           HDIM, DKH, 1, HDIM, DKH, T, 1, (long long)T * T, 0, 0, 0, 0, 0, NHEAD);
      softmax_kernel<<<NHEAD * T, 256, 0, stream>>>(Sbuf, T, 0.1020620726159658f); // 1/sqrt(96)
      // Z[h] = S[h] * V[h]
      gemm(Sbuf, vp, nullptr, nullptr, zp, T, DKH, T,
           T, (long long)T * T, HDIM, 1, DKH, HDIM, 1, DKH, 0, 0, 0, 0, 0, NHEAD);
    }
    // x3 = Z @ Wo + bo + x
    gemm(zb, Wo, bo, x, x3, BT, D_MODEL, HDIM,
         HDIM, 0, D_MODEL, 1, 0, D_MODEL, 1, 0, 0, 0, 0, 0, 0, 1);
    // x4 = LN(x3) -> x1
    ln_kernel<<<(BT + 7) / 8, 256, 0, stream>>>(x3, nullptr, ln2g, ln2b, x1, BT, D_MODEL);
    // conv1 (SAME, K=3) as 3 shifted GEMMs; relu on last
    for (int kk = 0; kk < 3; ++kk)
      gemm(x1, c1w + kk, kk == 0 ? c1b : nullptr, nullptr, hbuf, BT, CHID, D_MODEL,
           D_MODEL, 0, 3, (long long)D_MODEL * 3, 0, CHID, 1, 0, 0,
           kk - 1, T, kk > 0, kk == 2 ? 1 : 0, 1);
    // conv2
    for (int kk = 0; kk < 3; ++kk)
      gemm(hbuf, c2w + kk, kk == 0 ? c2b : nullptr, nullptr, h2, BT, D_MODEL, CHID,
           CHID, 0, 3, (long long)CHID * 3, 0, D_MODEL, 1, 0, 0,
           kk - 1, T, kk > 0, kk == 2 ? 1 : 0, 1);
    // x = LN(h2 + x3)
    ln_kernel<<<(BT + 7) / 8, 256, 0, stream>>>(h2, x3, ln3g, ln3b, x, BT, D_MODEL);
  };

  // ---- encoder ----
  embed_pe_kernel<<<BATCH * LPTOK, 128, 0, stream>>>(tokens, emb, xe, LPTOK, D_MODEL);
  for (int i = 0; i < 6; ++i) fft_block(xe, BATCH, LPTOK, ENC0 + i * 18);

  // ---- length regulate + decoder PE ----
  length_reg_kernel<<<BATCH, 256, 0, stream>>>(xe, durs, tlens, mel, LPTOK, TMELS, D_MODEL);

  // ---- decoder ----
  for (int i = 0; i < 6; ++i) fft_block(mel, BATCH, TMELS, DEC0 + i * 18);

  // ---- final projection with transposed store: out[b, n, t] ----
  gemm(mel, P(OWI), P(OBI), nullptr, (float*)d_out, TMELS, NMELS, D_MODEL,
       D_MODEL, (long long)TMELS * D_MODEL, NMELS, 1, 0,
       1, TMELS, (long long)NMELS * TMELS, 0, 0, 0, 0, 0, BATCH);
}
